// Attention_19146964205962
// MI455X (gfx1250) — matrix operand order repaired and checked by reading, not compiled, not run on hardware
//
#include <hip/hip_runtime.h>
#include <cstdint>

// ---------------------------------------------------------------------------
// MI455X (gfx1250) fused attention:
//   LN -> QKV gemm (bf16 WMMA) -> flash attention (bf16 WMMA, async-LDS
//   double-buffered K/V tiles) -> out proj.
// ~110 GFLOP vs ~50MB HBM traffic => matrix-compute bound; everything heavy
// goes through v_wmma_f32_16x16x32_bf16 with f32 accumulation.
// ---------------------------------------------------------------------------

#define BATCH   2
#define SEQ     2048
#define DMODEL  1024
#define HEADS   16
#define DH      64
#define INNER   (HEADS * DH)     // 1024
#define NNULL   2
#define NKEYS   (NNULL + SEQ)    // 2050
#define NPAD    2112             // 33 key tiles of 64
#define BN      (BATCH * SEQ)    // 4096 rows
#define EPS     1e-5f
#define NEGBIG  (-3.0e38f)

typedef __attribute__((ext_vector_type(16))) __bf16 v16bf;
typedef __attribute__((ext_vector_type(8)))  __bf16 v8bf;
typedef __attribute__((ext_vector_type(8)))  float  v8f;

static __device__ __forceinline__ v8f wmma_bf16(v16bf a, v16bf b, v8f c) {
    // (neg_a, A, neg_b, B, c_mod, C, reuse_a, reuse_b)
    return __builtin_amdgcn_wmma_f32_16x16x32_bf16(false, a, false, b,
                                                   (short)0, c, false, false);
}

// Build a v16bf fragment from two 8-halfword (16B) chunks.
static __device__ __forceinline__ v16bf frag16(const __bf16* p0, const __bf16* p1) {
    v8bf lo = *(const v8bf*)p0;
    v8bf hi = *(const v8bf*)p1;
    v16bf r;
#pragma unroll
    for (int e = 0; e < 8; ++e) { r[e] = lo[e]; r[e + 8] = hi[e]; }
    return r;
}

static __device__ __forceinline__ v8f zero8() {
    v8f z = {0.f, 0.f, 0.f, 0.f, 0.f, 0.f, 0.f, 0.f};
    return z;
}

// Async global->LDS 16B copy (CDNA5 VGLOBAL async encoding, ASYNCcnt-tracked).
// VDST holds the per-lane LDS byte address (low 32 bits of the flat address).
static __device__ __forceinline__ void async_ld16(void* lds_ptr, const void* gptr) {
    uint32_t lds_off = (uint32_t)(uintptr_t)lds_ptr;
    asm volatile("global_load_async_to_lds_b128 %0, %1, off"
                 :: "v"(lds_off), "v"(gptr)
                 : "memory");
}
static __device__ __forceinline__ void wait_async_le8() {
    asm volatile("s_wait_asynccnt 0x8" ::: "memory");
}
static __device__ __forceinline__ void wait_async_0() {
    asm volatile("s_wait_asynccnt 0x0" ::: "memory");
}

// ---------------------------------------------------------------------------
// 1) f32 -> bf16 weight conversion WITH transpose: dst[c*K + k] = src[k*NC + c]
//    so GEMM B-fragments are two contiguous per-lane 8-element (16B) runs (no LDS).
// ---------------------------------------------------------------------------
__global__ void cvt_transpose_kernel(const float* __restrict__ src,
                                     __bf16* __restrict__ dst, int K, int NC) {
    int i = blockIdx.x * blockDim.x + threadIdx.x;
    const int n = K * NC;
    const int stride = gridDim.x * blockDim.x;
    for (; i < n; i += stride) {
        int c = i / K;
        int k = i - c * K;
        dst[i] = (__bf16)src[(size_t)k * NC + c];
    }
}

// ---------------------------------------------------------------------------
// 2) LayerNorm over D=1024, one 256-thread block per row, bf16 output.
// ---------------------------------------------------------------------------
__global__ void layernorm_kernel(const float* __restrict__ x,
                                 const float* __restrict__ gamma,
                                 const float* __restrict__ beta,
                                 __bf16* __restrict__ xn) {
    const int row = blockIdx.x;
    const int tid = threadIdx.x;
    const float* xr = x + (size_t)row * DMODEL;

    float s = 0.f, ss = 0.f;
#pragma unroll
    for (int i = tid; i < DMODEL; i += 256) {
        float v = xr[i];
        s += v; ss += v * v;
    }
#pragma unroll
    for (int off = 16; off > 0; off >>= 1) {
        s  += __shfl_xor(s,  off, 32);
        ss += __shfl_xor(ss, off, 32);
    }
    __shared__ float red0[8], red1[8];
    __shared__ float mu_s, rs_s;
    if ((tid & 31) == 0) { red0[tid >> 5] = s; red1[tid >> 5] = ss; }
    __syncthreads();
    if (tid == 0) {
        float a = 0.f, b2 = 0.f;
#pragma unroll
        for (int j = 0; j < 8; ++j) { a += red0[j]; b2 += red1[j]; }
        float mu  = a / (float)DMODEL;
        float var = b2 / (float)DMODEL - mu * mu;
        mu_s = mu;
        rs_s = rsqrtf(var + EPS);
    }
    __syncthreads();
    const float mu = mu_s, rs = rs_s;
#pragma unroll
    for (int i = tid; i < DMODEL; i += 256) {
        xn[(size_t)row * DMODEL + i] = (__bf16)((xr[i] - mu) * rs * gamma[i] + beta[i]);
    }
}

// ---------------------------------------------------------------------------
// 3) Null-KV prefix rows + zero tail padding of the key dimension.
// ---------------------------------------------------------------------------
__global__ void fill_nullkv_kernel(const float* __restrict__ null_kv,
                                   __bf16* __restrict__ k_bf,
                                   __bf16* __restrict__ vT_bf) {
    int idx = blockIdx.x * blockDim.x + threadIdx.x;
    const int total = BATCH * HEADS * NPAD * DH;
    if (idx >= total) return;
    const int dh  = idx & (DH - 1);
    const int row = (idx >> 6) % NPAD;
    const int bh  = idx / (NPAD * DH);
    const int h   = bh % HEADS;
    if (row < NNULL) {
        float kv = null_kv[((size_t)h * 2 * NNULL + 2 * row)     * DH + dh];
        float vv = null_kv[((size_t)h * 2 * NNULL + 2 * row + 1) * DH + dh];
        k_bf [((size_t)bh * NPAD + row) * DH + dh]  = (__bf16)kv;
        vT_bf[((size_t)bh * DH + dh) * NPAD + row]  = (__bf16)vv;
    } else if (row >= NKEYS) {
        k_bf [((size_t)bh * NPAD + row) * DH + dh]  = (__bf16)0.0f;
        vT_bf[((size_t)bh * DH + dh) * NPAD + row]  = (__bf16)0.0f;
    }
}

// ---------------------------------------------------------------------------
// Tiled bf16 WMMA GEMM:  C[M,Ncols] = A[M,K] x BT^T  with BT = B^T [Ncols,K].
// Block = 128 thr (4 waves), tile 64x64, K-step 32.  No LDS: both A and B
// fragments are contiguous per-lane 16B loads (weights L2-resident).
// EPILOGUE 0: route to q / k / vT (QKV).  EPILOGUE 1: f32 store (out proj).
// ---------------------------------------------------------------------------
template <int EPILOGUE>
__global__ void wmma_gemm_kernel(const __bf16* __restrict__ A,
                                 const __bf16* __restrict__ BT,
                                 int M, int Ncols, int K,
                                 __bf16* __restrict__ q_bf,
                                 __bf16* __restrict__ k_bf,
                                 __bf16* __restrict__ vT_bf,
                                 float* __restrict__ out) {
    const int tid  = threadIdx.x;
    const int wave = tid >> 5;
    const int lane = tid & 31;
    const int half = lane >> 4;
    const int lq   = lane & 15;

    const int row0 = blockIdx.y * 64 + wave * 16;
    const int col0 = blockIdx.x * 64;
    const int m    = row0 + lq;

    v8f acc[4];
#pragma unroll
    for (int c = 0; c < 4; ++c) acc[c] = zero8();

    for (int k0 = 0; k0 < K; k0 += 32) {
        const __bf16* arow = &A[(size_t)m * K + k0];
        v16bf afrag = frag16(arow + 8 * half, arow + 16 + 8 * half);
        if (k0 + 32 < K) {
            __builtin_prefetch(arow + 32, 0, 0);
            __builtin_prefetch(&BT[(size_t)(col0 + 16 + lq) * K + k0 + 32], 0, 0);
        }
#pragma unroll
        for (int c = 0; c < 4; ++c) {
            const __bf16* bcol = &BT[(size_t)(col0 + 16 * c + lq) * K + k0];
            v16bf bfrag = frag16(bcol + 8 * half, bcol + 16 + 8 * half);
            acc[c] = wmma_bf16(afrag, bfrag, acc[c]);
        }
    }

    // Epilogue: C layout -> lane lq = column-in-subtile, VGPR r = row (+8*half)
#pragma unroll
    for (int c = 0; c < 4; ++c) {
#pragma unroll
        for (int r = 0; r < 8; ++r) {
            const int mm = row0 + r + 8 * half;
            const int cg = col0 + 16 * c + lq;
            const float val = acc[c][r];
            if (EPILOGUE == 0) {
                const int which  = cg >> 10;         // 0=q 1=k 2=v
                const int within = cg & 1023;
                const int h  = within >> 6;
                const int dh = within & 63;
                const int b  = mm >> 11;             // / SEQ
                const int n  = mm & 2047;            // % SEQ
                const size_t bh = (size_t)b * HEADS + h;
                if (which == 0)
                    q_bf[(bh * SEQ + n) * DH + dh] = (__bf16)(val * 0.125f); // DH^-0.5
                else if (which == 1)
                    k_bf[(bh * NPAD + (NNULL + n)) * DH + dh] = (__bf16)val;
                else
                    vT_bf[(bh * DH + dh) * NPAD + (NNULL + n)] = (__bf16)val;
            } else {
                out[(size_t)mm * Ncols + cg] = val;
            }
        }
    }
}

// ---------------------------------------------------------------------------
// 4) Flash attention.  Grid (SEQ/64, B*H); 128 threads = 4 waves.
//    K/V 64x64 tiles double-buffered in LDS via global_load_async_to_lds_b128
//    (shared by all 4 waves -> 4x less global traffic, latency hidden).
//    Row sums come free from an extra P @ ones WMMA (no shuffle reduction).
// ---------------------------------------------------------------------------
struct KVTiles {
    __bf16 k[2][64][64];
    __bf16 v[2][64][64];
};

static __device__ __forceinline__ void stage_kv(KVTiles* t, int buf, int j0,
                                                int bh, int tid,
                                                const __bf16* __restrict__ k_bf,
                                                const __bf16* __restrict__ vT_bf) {
    // K tile: 64 rows x 128B = 512 chunks of 16B; 128 threads x 4 chunks.
#pragma unroll
    for (int i = 0; i < 4; ++i) {
        const int c   = tid + i * 128;
        const int row = c >> 3;
        const int off = (c & 7) * 8;
        async_ld16(&t->k[buf][row][off],
                   k_bf + ((size_t)bh * NPAD + j0 + row) * DH + off);
    }
    // V^T tile: 64 dim-rows x 64 keys.
#pragma unroll
    for (int i = 0; i < 4; ++i) {
        const int c   = tid + i * 128;
        const int row = c >> 3;
        const int off = (c & 7) * 8;
        async_ld16(&t->v[buf][row][off],
                   vT_bf + ((size_t)bh * DH + row) * NPAD + j0 + off);
    }
}

__global__ void attention_kernel(const __bf16* __restrict__ q_bf,
                                 const __bf16* __restrict__ k_bf,
                                 const __bf16* __restrict__ vT_bf,
                                 const unsigned char* __restrict__ mask,
                                 __bf16* __restrict__ ao_bf) {
    __shared__ __align__(16) KVTiles tiles;                 // 32 KB
    __shared__ __align__(16) __bf16 lds_p[4][16][64];       // 8 KB

    const int bh   = blockIdx.y;
    const int b    = bh / HEADS;
    const int h    = bh % HEADS;
    const int tid  = threadIdx.x;
    const int wave = tid >> 5;
    const int lane = tid & 31;
    const int half = lane >> 4;
    const int lq   = lane & 15;

    const int qrow0 = blockIdx.x * 64 + wave * 16;

    // Preload Q A-fragments (head dim 64 -> two 16x32 frags).
    const __bf16* qrow = q_bf + ((size_t)bh * SEQ + qrow0 + lq) * DH;
    const v16bf qa0 = frag16(qrow +      8 * half, qrow + 16 + 8 * half);
    const v16bf qa1 = frag16(qrow + 32 + 8 * half, qrow + 48 + 8 * half);

    // All-ones B fragment: P @ ones replicates row sums into every lane.
    v16bf ones;
#pragma unroll
    for (int e = 0; e < 16; ++e) ones[e] = (__bf16)1.0f;

    v8f o[4], osum;
#pragma unroll
    for (int d = 0; d < 4; ++d) o[d] = zero8();
    osum = zero8();
    float m_i[8];
#pragma unroll
    for (int r = 0; r < 8; ++r) m_i[r] = NEGBIG;

    // Prime the async pipeline with tile 0.
    stage_kv(&tiles, 0, 0, bh, tid, k_bf, vT_bf);

    int buf = 0;
    for (int j0 = 0; j0 < NPAD; j0 += 64, buf ^= 1) {
        const bool has_next = (j0 + 64 < NPAD);
        if (has_next) {
            __syncthreads();                    // prior reads of buf^1 finished
            stage_kv(&tiles, buf ^ 1, j0 + 64, bh, tid, k_bf, vT_bf);
            wait_async_le8();                   // current tile's 8 loads done
        } else {
            wait_async_0();
        }
        __syncthreads();                        // tile visible to all waves

        // ---- S = Q @ K^T (16 x 64 scores) ----
        v8f s[4];
#pragma unroll
        for (int t = 0; t < 4; ++t) s[t] = zero8();
#pragma unroll
        for (int t = 0; t < 4; ++t) {
            const __bf16* krow = &tiles.k[buf][16 * t + lq][0];
            v16bf kb0 = frag16(krow +      8 * half, krow + 16 + 8 * half);
            v16bf kb1 = frag16(krow + 32 + 8 * half, krow + 48 + 8 * half);
            s[t] = wmma_bf16(qa0, kb0, s[t]);
            s[t] = wmma_bf16(qa1, kb1, s[t]);
        }
        // ---- mask (null prefix valid; tail padding invalid) ----
#pragma unroll
        for (int t = 0; t < 4; ++t) {
            const int key = j0 + 16 * t + lq;
            bool valid = (key < NNULL) ||
                         (key < NKEYS && mask[(size_t)b * SEQ + (key - NNULL)] != 0);
            if (!valid) {
#pragma unroll
                for (int r = 0; r < 8; ++r) s[t][r] = NEGBIG;
            }
        }
        // ---- row max across the 64-key tile ----
        float tmax[8];
#pragma unroll
        for (int r = 0; r < 8; ++r) {
            float v = s[0][r];
#pragma unroll
            for (int t = 1; t < 4; ++t) v = fmaxf(v, s[t][r]);
            tmax[r] = v;
        }
#pragma unroll
        for (int off = 1; off < 16; off <<= 1) {
#pragma unroll
            for (int r = 0; r < 8; ++r)
                tmax[r] = fmaxf(tmax[r], __shfl_xor(tmax[r], off, 32));
        }
        // ---- online softmax rescale ----
        float m_new[8], alpha[8];
#pragma unroll
        for (int r = 0; r < 8; ++r) {
            m_new[r] = fmaxf(m_i[r], tmax[r]);
            alpha[r] = __expf(m_i[r] - m_new[r]);
            m_i[r]   = m_new[r];
        }
#pragma unroll
        for (int t = 0; t < 4; ++t)
#pragma unroll
            for (int r = 0; r < 8; ++r) s[t][r] = __expf(s[t][r] - m_new[r]);
#pragma unroll
        for (int d = 0; d < 4; ++d)
#pragma unroll
            for (int r = 0; r < 8; ++r) o[d][r] = o[d][r] * alpha[r];
#pragma unroll
        for (int r = 0; r < 8; ++r) osum[r] = osum[r] * alpha[r];

        // ---- stage P (C layout) to LDS, reload in A layout ----
#pragma unroll
        for (int t = 0; t < 4; ++t)
#pragma unroll
            for (int r = 0; r < 8; ++r)
                lds_p[wave][r + 8 * half][16 * t + lq] = (__bf16)s[t][r];

        const __bf16* prow = &lds_p[wave][lq][0];
        v16bf pa0 = frag16(prow +      8 * half, prow + 16 + 8 * half);
        v16bf pa1 = frag16(prow + 32 + 8 * half, prow + 48 + 8 * half);

        // ---- O += P @ V ; row-sum via P @ ones ----
#pragma unroll
        for (int d = 0; d < 4; ++d) {
            const __bf16* vrow = &tiles.v[buf][16 * d + lq][0];
            v16bf vb0 = frag16(vrow +      8 * half, vrow + 16 + 8 * half);
            v16bf vb1 = frag16(vrow + 32 + 8 * half, vrow + 48 + 8 * half);
            o[d] = wmma_bf16(pa0, vb0, o[d]);
            o[d] = wmma_bf16(pa1, vb1, o[d]);
        }
        osum = wmma_bf16(pa0, ones, osum);
        osum = wmma_bf16(pa1, ones, osum);
    }

    // ---- normalize and store [B, N, H*DH] bf16 ----
#pragma unroll
    for (int d = 0; d < 4; ++d) {
#pragma unroll
        for (int r = 0; r < 8; ++r) {
            const int row = qrow0 + r + 8 * half;
            const float val = o[d][r] / osum[r];
            ao_bf[((size_t)b * SEQ + row) * INNER + h * DH + 16 * d + lq] = (__bf16)val;
        }
    }
}

// ---------------------------------------------------------------------------
// Host launcher
// ---------------------------------------------------------------------------
extern "C" void kernel_launch(void* const* d_in, const int* in_sizes, int n_in,
                              void* d_out, int out_size, void* d_ws, size_t ws_size,
                              hipStream_t stream) {
    (void)in_sizes; (void)n_in; (void)out_size; (void)ws_size;
    const float*         x       = (const float*)d_in[0];
    const unsigned char* mask    = (const unsigned char*)d_in[1];
    const float*         gamma   = (const float*)d_in[2];
    const float*         beta    = (const float*)d_in[3];
    const float*         null_kv = (const float*)d_in[4];
    const float*         w_qkv   = (const float*)d_in[5];
    const float*         w_out   = (const float*)d_in[6];
    float*               out     = (float*)d_out;

    char* p = (char*)d_ws;
    __bf16* xn      = (__bf16*)p; p += (size_t)BN * DMODEL        * sizeof(__bf16);
    __bf16* wqkvT   = (__bf16*)p; p += (size_t)DMODEL * 3 * INNER * sizeof(__bf16);
    __bf16* woutT   = (__bf16*)p; p += (size_t)INNER * DMODEL     * sizeof(__bf16);
    __bf16* q_bf    = (__bf16*)p; p += (size_t)BATCH * HEADS * SEQ  * DH * sizeof(__bf16);
    __bf16* k_bf    = (__bf16*)p; p += (size_t)BATCH * HEADS * NPAD * DH * sizeof(__bf16);
    __bf16* vT_bf   = (__bf16*)p; p += (size_t)BATCH * HEADS * DH * NPAD * sizeof(__bf16);
    __bf16* ao_bf   = (__bf16*)p; p += (size_t)BN * INNER         * sizeof(__bf16);

    // 1) transposed weight conversion (BT[col][k])
    {
        const int n1 = DMODEL * 3 * INNER;
        const int n2 = INNER * DMODEL;
        cvt_transpose_kernel<<<(n1 + 1023) / 1024, 256, 0, stream>>>(w_qkv, wqkvT,
                                                                     DMODEL, 3 * INNER);
        cvt_transpose_kernel<<<(n2 + 1023) / 1024, 256, 0, stream>>>(w_out, woutT,
                                                                     INNER, DMODEL);
    }
    // 2) layernorm
    layernorm_kernel<<<BN, 256, 0, stream>>>(x, gamma, beta, xn);
    // 3) null-kv prefix + tail padding
    {
        const int tot = BATCH * HEADS * NPAD * DH;
        fill_nullkv_kernel<<<(tot + 255) / 256, 256, 0, stream>>>(null_kv, k_bf, vT_bf);
    }
    // 4) QKV projection (bf16 WMMA) with routing epilogue
    wmma_gemm_kernel<0><<<dim3((3 * INNER) / 64, BN / 64), 128, 0, stream>>>(
        xn, wqkvT, BN, 3 * INNER, DMODEL, q_bf, k_bf, vT_bf, nullptr);
    // 5) flash attention (bf16 WMMA + async-LDS K/V staging)
    attention_kernel<<<dim3(SEQ / 64, BATCH * HEADS), 128, 0, stream>>>(
        q_bf, k_bf, vT_bf, mask, ao_bf);
    // 6) output projection (bf16 WMMA, f32 out)
    wmma_gemm_kernel<1><<<dim3(DMODEL / 64, BN / 64), 128, 0, stream>>>(
        ao_bf, woutT, BN, DMODEL, INNER, nullptr, nullptr, nullptr, out);
}
